// Siamese_HingeCheby_70849780514835
// MI455X (gfx1250) — compile-verified, matrix operand order checked by reading, not run on hardware
//
#include <hip/hip_runtime.h>
#include <hip/hip_bf16.h>
#include <stdint.h>

// ---------------------------------------------------------------------------
// Types for CDNA5 WMMA (wave32): v_wmma_f32_16x16x32_bf16
// ---------------------------------------------------------------------------
typedef __attribute__((ext_vector_type(16))) __bf16       v16bf;
typedef __attribute__((ext_vector_type(16))) short        v16s;
typedef __attribute__((ext_vector_type(8)))  float        v8f;
typedef __attribute__((ext_vector_type(4)))  unsigned int v4u;
typedef __attribute__((ext_vector_type(8)))  int          v8i;
typedef __attribute__((ext_vector_type(4)))  int          v4i;

union FragAB {
    v16s  s;
    v16bf b;
    uint4 u4[2];
};

__device__ __forceinline__ unsigned short f32_to_bf16_rne(float f) {
    union { float f; unsigned int u; } x;
    x.f = f;
    unsigned int u = x.u;
    unsigned int r = u + 0x7FFFu + ((u >> 16) & 1u);  // round-to-nearest-even
    return (unsigned short)(r >> 16);
}

// ---------------------------------------------------------------------------
// Tensor Data Mover: async-load a 2D tile of f32 weights (tile_dim0=16 cols,
// tile_dim1=32 rows, row stride Nd) into LDS.  D# built per CDNA5 ISA §8:
//   group0: count=1 | lds_addr | global_addr[56:0] | type=2
//   group1: data_size=4B | tensor_dim0=Nd | tensor_dim1=Kd | tile 16x32 |
//           tensor_dim0_stride=Nd
// Tracked with TENSORcnt; completion enforced via s_wait_tensorcnt.
// ---------------------------------------------------------------------------
__device__ __forceinline__ void tdm_load_b_tile(
    unsigned lds_addr, const float* __restrict__ B, int Nd, int Kd,
    int kb, int ntile)
{
    const float* g = B + (size_t)kb * Nd + (size_t)ntile * 16;
    unsigned long long ga = (unsigned long long)(uintptr_t)g;
    unsigned ga_lo = (unsigned)ga;
    unsigned ga_hi = (unsigned)(ga >> 32);

    v4u g0;
    g0.x = 1u;                                   // count=1, user desc, no gather
    g0.y = lds_addr;                             // LDS byte address of tile
    g0.z = ga_lo;                                // global_addr[31:0]
    g0.w = (ga_hi & 0x01FFFFFFu) | (2u << 30);   // global_addr[56:32] | type=2

    v8i g1;
    g1[0] = (int)(2u << 16);                                         // data_size=4B
    g1[1] = (int)(((unsigned)Nd & 0xFFFFu) << 16);                   // tensor_dim0 lo16
    g1[2] = (int)((((unsigned)Nd >> 16) & 0xFFFFu) |
                  (((unsigned)Kd & 0xFFFFu) << 16));                 // dim0 hi | dim1 lo
    g1[3] = (int)((((unsigned)Kd >> 16) & 0xFFFFu) | (16u << 16));   // dim1 hi | tile_dim0=16
    g1[4] = 32;                                                      // tile_dim1=32, tile_dim2=0
    g1[5] = Nd;                                                      // tensor_dim0_stride lo32
    g1[6] = 0;                                                       // stride hi | dim1_stride lo
    g1[7] = 0;

    v4i z4 = {0, 0, 0, 0};
#if defined(__clang_major__) && (__clang_major__ >= 23)
    v8i z8 = {0, 0, 0, 0, 0, 0, 0, 0};
    __builtin_amdgcn_tensor_load_to_lds(g0, g1, z4, z4, z8, 0);
#else
    __builtin_amdgcn_tensor_load_to_lds(g0, g1, z4, z4, 0);
#endif
}

// ---------------------------------------------------------------------------
// Edge normalization: nw[e] = dinv[src]*ew[e]*dinv[dst], dinv = rsqrt(deg)
// ---------------------------------------------------------------------------
__global__ __launch_bounds__(256) void compute_nw_kernel(
    const int* __restrict__ ei, const float* __restrict__ ew,
    float* __restrict__ nw, int E, int Nn)
{
    __shared__ float deg[256];
    const int tid = threadIdx.x;
    const int* src = ei;
    const int* dst = ei + E;
    for (int i = tid; i < 256; i += blockDim.x) deg[i] = 0.f;
    __syncthreads();
    for (int e = tid; e < E; e += blockDim.x) atomicAdd(&deg[src[e]], ew[e]);
    __syncthreads();
    for (int i = tid; i < Nn; i += blockDim.x) {
        float d = deg[i];
        deg[i] = (d > 0.f) ? rsqrtf(d) : 0.f;
    }
    __syncthreads();
    for (int e = tid; e < E; e += blockDim.x)
        nw[e] = deg[src[e]] * ew[e] * deg[dst[e]];
}

__global__ __launch_bounds__(256) void zero_f32_kernel(float* __restrict__ p, size_t n)
{
    size_t i = (size_t)blockIdx.x * blockDim.x + threadIdx.x;
    if (i < n) p[i] = 0.f;
}

// out[dst[e], :] += -nw[e] * t[src[e], :]   (coalesced over features)
__global__ __launch_bounds__(256) void prop_kernel(
    const float* __restrict__ nw, const int* __restrict__ src,
    const int* __restrict__ dst, const float* __restrict__ t,
    float* __restrict__ out, int F)
{
    const int e = blockIdx.y;
    const int f = blockIdx.x * blockDim.x + threadIdx.x;
    const float w = -nw[e];
    const int s = src[e];
    const int d = dst[e];
    atomicAdd(&out[(size_t)d * F + f], w * t[(size_t)s * F + f]);
}

// t2 = 2*t2 - x on valid rows; pad rows -> 0 (in place, padded [PR, F])
__global__ __launch_bounds__(256) void finish_t2_kernel(
    float* __restrict__ t2, const float* __restrict__ x,
    int F, int validM, int PR)
{
    size_t i = (size_t)blockIdx.x * blockDim.x + threadIdx.x;
    size_t total = (size_t)PR * F;
    if (i >= total) return;
    int row = (int)(i / (size_t)F);
    float v = 0.f;
    if (row < validM) v = 2.f * t2[i] - x[i];
    t2[i] = v;
}

// f32 -> bf16 with zero padding of rows >= validM (dst padded [PRA, F])
__global__ __launch_bounds__(256) void cvt_bf16_pad_kernel(
    unsigned short* __restrict__ dstp, const float* __restrict__ srcp,
    int F, int validM, int PRA)
{
    size_t i = (size_t)blockIdx.x * blockDim.x + threadIdx.x;
    size_t total = (size_t)PRA * F;
    if (i >= total) return;
    int row = (int)(i / (size_t)F);
    unsigned short v = 0;
    if (row < validM) v = f32_to_bf16_rne(srcp[i]);
    dstp[i] = v;
}

// B fragment from the LDS-staged 32x16 f32 tile (row r at offset r*16 floats):
// b[j]: K = (j>>3)*16 + half*8 + (j&7), column = nloc.
__device__ __forceinline__ v16bf load_b_frag_lds(
    const float* __restrict__ lbuf, int nloc, int half)
{
    FragAB t;
#pragma unroll
    for (int j = 0; j < 16; ++j) {
        int k = (j >> 3) * 16 + half * 8 + (j & 7);
        t.s[j] = (short)f32_to_bf16_rne(lbuf[k * 16 + nloc]);
    }
    return t.b;
}

// ---------------------------------------------------------------------------
// ChebConv GEMM:  out[., Nd] = A0@B0 + A1@B1 + A2@B2 + bias, ReLU.
// A*: bf16 [256, Kd] row-major (zero-padded).  B*: f32 [Kd, Nd] row-major.
// 4 waves/block; each wave owns one 16-wide N strip and ALL 16 M-tiles, so
// every weight element is read from HBM exactly once.  B tiles are streamed
// with TENSOR_LOAD_TO_LDS, double-buffered per wave, synced via TENSORcnt.
// No branches in the unrolled M-tile loop -> clean back-to-back WMMA issue.
// ---------------------------------------------------------------------------
__global__ __launch_bounds__(128) void gemm_cheb_tdm_kernel(
    const unsigned short* __restrict__ A0, const unsigned short* __restrict__ A1,
    const unsigned short* __restrict__ A2,
    const float* __restrict__ B0, const float* __restrict__ B1,
    const float* __restrict__ B2,
    const float* __restrict__ bias, float* __restrict__ out,
    int Kd, int Nd, int validM)
{
    __shared__ float bsm[4][2][512];     // per-wave double-buffered 32x16 f32 tile

    const int wave  = __builtin_amdgcn_readfirstlane((int)(threadIdx.x >> 5));
    const int lane  = threadIdx.x & 31;
    const int half  = lane >> 4;
    const int nloc  = lane & 15;
    const int ntile = __builtin_amdgcn_readfirstlane((int)(blockIdx.x * 4) + wave);
    const int col   = ntile * 16 + nloc;

    unsigned lds_addr[2];
    lds_addr[0] = __builtin_amdgcn_readfirstlane(
        (int)(unsigned)(uintptr_t)(void*)&bsm[wave][0][0]);
    lds_addr[1] = __builtin_amdgcn_readfirstlane(
        (int)(unsigned)(uintptr_t)(void*)&bsm[wave][1][0]);

    v8f acc[16];
#pragma unroll
    for (int i = 0; i < 16; ++i)
#pragma unroll
        for (int v = 0; v < 8; ++v) acc[i][v] = 0.f;

    const unsigned short* As[3] = {A0, A1, A2};
    const float*          Bs[3] = {B0, B1, B2};

#pragma unroll
    for (int kk = 0; kk < 3; ++kk) {
        const unsigned short* __restrict__ A = As[kk];
        const float* __restrict__ B = Bs[kk];

        tdm_load_b_tile(lds_addr[0], B, Nd, Kd, 0, ntile);   // prologue
        int pp = 0;
        for (int kb = 0; kb < Kd; kb += 32) {
            if (kb + 32 < Kd) {
                tdm_load_b_tile(lds_addr[pp ^ 1], B, Nd, Kd, kb + 32, ntile);
                __builtin_amdgcn_s_wait_tensorcnt(1);  // current tile landed
            } else {
                __builtin_amdgcn_s_wait_tensorcnt(0);
            }

            v16bf bf = load_b_frag_lds(&bsm[wave][pp][0], nloc, half);

            // A frag base for this lane: row m = nloc, K offsets half*8 / 16+half*8
            const unsigned short* Ak = A + (size_t)nloc * Kd + kb + half * 8;
#pragma unroll
            for (int mi = 0; mi < 16; ++mi) {
                const uint4* pa = (const uint4*)(Ak + (size_t)mi * 16 * Kd);
                FragAB t;
                t.u4[0] = pa[0];   // K = kb + half*8 + 0..7
                t.u4[1] = pa[2];   // K = kb + 16 + half*8 + 0..7
                acc[mi] = __builtin_amdgcn_wmma_f32_16x16x32_bf16(
                    false, t.b, false, bf, (short)0, acc[mi], false, false);
            }
            pp ^= 1;
        }
    }

    const float bv = bias[col];
#pragma unroll
    for (int mi = 0; mi < 16; ++mi) {
#pragma unroll
        for (int v = 0; v < 8; ++v) {
            const int row = mi * 16 + half * 8 + v;   // C/D layout: M = half*8 + v
            if (row < validM) {
                float r = acc[mi][v] + bv;
                out[(size_t)row * Nd + col] = (r > 0.f) ? r : 0.f;
            }
        }
    }
}

// ---------------------------------------------------------------------------
// Classifier: out[c] = sum_j relu( sum_n (o1[n,c]*o2[n,c]) w1[n,j] + b1[j] ) w2[j] + b2
// ---------------------------------------------------------------------------
__global__ __launch_bounds__(128) void classifier_kernel(
    const float* __restrict__ o1, const float* __restrict__ o2,
    const float* __restrict__ w1, const float* __restrict__ b1,
    const float* __restrict__ w2, const float* __restrict__ b2,
    float* __restrict__ out, int Nn, int NC)
{
    __shared__ float p[256];
    __shared__ float red[128];
    const int c   = blockIdx.x;
    const int tid = threadIdx.x;

    for (int n = tid; n < Nn; n += blockDim.x)
        p[n] = o1[(size_t)n * NC + c] * o2[(size_t)n * NC + c];
    __syncthreads();

    float part = 0.f;
    for (int j = tid; j < 100; j += blockDim.x) {
        float hj = b1[j];
        for (int n = 0; n < Nn; ++n) hj += p[n] * w1[n * 100 + j];
        hj = (hj > 0.f) ? hj : 0.f;
        part += hj * w2[j];
    }
    red[tid] = part;
    __syncthreads();
    for (int s = 64; s > 0; s >>= 1) {
        if (tid < s) red[tid] += red[tid + s];
        __syncthreads();
    }
    if (tid == 0) out[c] = red[0] + b2[0];
}

// ---------------------------------------------------------------------------
// Host orchestration
// ---------------------------------------------------------------------------
extern "C" void kernel_launch(void* const* d_in, const int* in_sizes, int n_in,
                              void* d_out, int out_size, void* d_ws, size_t ws_size,
                              hipStream_t stream)
{
    (void)in_sizes; (void)n_in; (void)out_size; (void)ws_size;

    const float* x[2]  = {(const float*)d_in[0], (const float*)d_in[1]};
    const int*   ei[2] = {(const int*)d_in[2],  (const int*)d_in[3]};
    const float* ea[2] = {(const float*)d_in[4], (const float*)d_in[5]};
    const float* W1  = (const float*)d_in[6];
    const float* b1  = (const float*)d_in[7];
    const float* W4  = (const float*)d_in[8];
    const float* b4  = (const float*)d_in[9];
    const float* l1w = (const float*)d_in[10];
    const float* l1b = (const float*)d_in[11];
    const float* l2w = (const float*)d_in[12];
    const float* l2b = (const float*)d_in[13];

    constexpr int Nn = 200, E = 3200;
    constexpr int F1 = 8192, F2 = 4096, NC = 4096;
    constexpr int PR  = 208;   // f32 scratch padding
    constexpr int PRA = 256;   // bf16 A padding: 16 WMMA M-tiles, no guards

    char* ws = (char*)d_ws;
    auto alloc = [&](size_t bytes) -> char* {
        char* p = ws;
        ws += (bytes + 255) & ~(size_t)255;
        return p;
    };
    float* nw          = (float*)alloc((size_t)E * 4);
    float* t1          = (float*)alloc((size_t)PR * F1 * 4);
    float* t2          = (float*)alloc((size_t)PR * F1 * 4);
    unsigned short* a0 = (unsigned short*)alloc((size_t)PRA * F1 * 2);
    unsigned short* a1 = (unsigned short*)alloc((size_t)PRA * F1 * 2);
    unsigned short* a2 = (unsigned short*)alloc((size_t)PRA * F1 * 2);
    float* h           = (float*)alloc((size_t)PR * F2 * 4);
    float* og[2];
    og[0]              = (float*)alloc((size_t)PR * NC * 4);
    og[1]              = (float*)alloc((size_t)PR * NC * 4);

    auto blocks_for = [](size_t n) { return (unsigned)((n + 255) / 256); };
    const size_t n1  = (size_t)PR  * F1;
    const size_t n2  = (size_t)PR  * F2;
    const size_t na1 = (size_t)PRA * F1;
    const size_t na2 = (size_t)PRA * F2;

    for (int g = 0; g < 2; ++g) {
        const int* src = ei[g];
        const int* dst = ei[g] + E;

        compute_nw_kernel<<<1, 256, 0, stream>>>(ei[g], ea[g], nw, E, Nn);

        // ---- layer 1: T0 = x, T1 = L x, T2 = 2 L T1 - x ----
        zero_f32_kernel<<<blocks_for(n1), 256, 0, stream>>>(t1, n1);
        prop_kernel<<<dim3(F1 / 256, E), 256, 0, stream>>>(nw, src, dst, x[g], t1, F1);
        zero_f32_kernel<<<blocks_for(n1), 256, 0, stream>>>(t2, n1);
        prop_kernel<<<dim3(F1 / 256, E), 256, 0, stream>>>(nw, src, dst, t1, t2, F1);
        finish_t2_kernel<<<blocks_for(n1), 256, 0, stream>>>(t2, x[g], F1, Nn, PR);

        cvt_bf16_pad_kernel<<<blocks_for(na1), 256, 0, stream>>>(a0, x[g], F1, Nn, PRA);
        cvt_bf16_pad_kernel<<<blocks_for(na1), 256, 0, stream>>>(a1, t1, F1, Nn, PRA);
        cvt_bf16_pad_kernel<<<blocks_for(na1), 256, 0, stream>>>(a2, t2, F1, Nn, PRA);

        gemm_cheb_tdm_kernel<<<F2 / 64, 128, 0, stream>>>(
            a0, a1, a2,
            W1, W1 + (size_t)F1 * F2, W1 + 2 * (size_t)F1 * F2,
            b1, h, F1, F2, Nn);

        // ---- layer 2 on h ----
        zero_f32_kernel<<<blocks_for(n2), 256, 0, stream>>>(t1, n2);
        prop_kernel<<<dim3(F2 / 256, E), 256, 0, stream>>>(nw, src, dst, h, t1, F2);
        zero_f32_kernel<<<blocks_for(n2), 256, 0, stream>>>(t2, n2);
        prop_kernel<<<dim3(F2 / 256, E), 256, 0, stream>>>(nw, src, dst, t1, t2, F2);
        finish_t2_kernel<<<blocks_for(n2), 256, 0, stream>>>(t2, h, F2, Nn, PR);

        cvt_bf16_pad_kernel<<<blocks_for(na2), 256, 0, stream>>>(a0, h, F2, Nn, PRA);
        cvt_bf16_pad_kernel<<<blocks_for(na2), 256, 0, stream>>>(a1, t1, F2, Nn, PRA);
        cvt_bf16_pad_kernel<<<blocks_for(na2), 256, 0, stream>>>(a2, t2, F2, Nn, PRA);

        gemm_cheb_tdm_kernel<<<NC / 64, 128, 0, stream>>>(
            a0, a1, a2,
            W4, W4 + (size_t)F2 * NC, W4 + 2 * (size_t)F2 * NC,
            b4, og[g], F2, NC, Nn);
    }

    classifier_kernel<<<NC, 128, 0, stream>>>(
        og[0], og[1], l1w, l1b, l2w, l2b, (float*)d_out, Nn, NC);
}
